// Net3_74088185856421
// MI455X (gfx1250) — matrix operand order repaired and checked by reading, not compiled, not run on hardware
//
#include <hip/hip_runtime.h>

// ============================================================================
// MLP (12 linear + 11 train-mode BatchNorm + 3 skip projections) over 1M rows.
//
// MI455X (gfx1250, wave32) design:
//  * BN couples the batch -> 12 stage kernels; stage K_i fuses
//    [bn_i + skip-add + relu] of the previous raw activation with
//    [linear_{i+1}] and fp32 accumulation of bn_{i+1} statistics.
//  * Activations ping-pong through two 80 MB fp16 buffers (fit in 192 MB L2),
//    rows padded to 8-half multiples so the WMMA A operand loads as two
//    aligned global_load_b128 runs per lane per K-chunk (padding masked by
//    zeroed BN coefficients / explicit select in skip stages).
//  * Matmuls: v_wmma_f32_16x16x32_f16 (fp32 accumulate), 16-row tile / wave;
//    weights pre-packed into registers in WMMA-B layout, bias baked into C.
//  * Loads/stores use one base pointer + compile-time element offsets so row
//    strides fold into the VMEM instruction's immediate offset field.
//  * Stats: per-lane regs -> LDS ds_add_f32 -> one global atomic per feature.
//  * global_prefetch_b8 of next grid-stride tile hides L2 latency.
// ============================================================================

#define BN_EPS 1e-5f

typedef __attribute__((ext_vector_type(16))) _Float16 v16h;
typedef __attribute__((ext_vector_type(8)))  _Float16 v8h;
typedef __attribute__((ext_vector_type(2)))  _Float16 v2h;
typedef __attribute__((ext_vector_type(8)))  float    v8f;

constexpr int BATCH = 1000000;
constexpr int TILES = BATCH / 16;   // 62500 16-row tiles
constexpr int BLOCK = 256;          // 8 wave32 waves
constexpr int WAVES = BLOCK / 32;
constexpr int GRID  = 1954;         // ~4 tiles per wave

constexpr int pad8(int f) { return (f + 7) & ~7; }

// ----------------------------------------------------------------------------
// Fused stage kernel.
//  FI / FO1 / FO2 : real feature counts (in, main out, optional skip out)
//  fp16 tensors use row stride pad8(F); padded tail is garbage and masked.
// ----------------------------------------------------------------------------
template <int FI, int FO1, int FO2, bool BN_IN, bool SKIP_IN, bool IN_F32>
__global__ __launch_bounds__(BLOCK) void stage_kernel(
    const void* __restrict__ zin_,
    const float* __restrict__ statsIn,            // sum[0..63], sumsq[64..127]
    const float* __restrict__ gam,
    const float* __restrict__ bet,
    const _Float16* __restrict__ skipIn,          // (BATCH, pad8(FI)) or null
    const float* __restrict__ W1, const float* __restrict__ B1,   // (FO1,FI)
    const float* __restrict__ W2, const float* __restrict__ B2,   // (FO2,FI)
    _Float16* __restrict__ zout1,                 // (BATCH, pad8(FO1))
    _Float16* __restrict__ zout2,                 // (BATCH, pad8(FO2))
    float* __restrict__ statsOut)                 // sum[0..63], sumsq[64..127]
{
    constexpr int KC   = (FI + 31) / 32;   // K chunks of 32
    constexpr int NT1  = (FO1 + 15) / 16;  // N tiles, main output
    constexpr int NT2  = (FO2 + 15) / 16;  // N tiles, skip output
    constexpr int NTT  = NT1 + NT2;
    constexpr int FIP  = pad8(FI);
    constexpr int FOP1 = pad8(FO1);
    constexpr int FOP2 = pad8(FO2 > 0 ? FO2 : 8);

    const float*    zf = (const float*)zin_;
    const _Float16* zh = (const _Float16*)zin_;

    const int lane = threadIdx.x & 31;
    const int wave = threadIdx.x >> 5;
    const int g    = lane >> 4;   // half-wave select
    const int nloc = lane & 15;

    __shared__ float ssScale[64];
    __shared__ float ssShift[64];
    __shared__ float sAcc[64];
    __shared__ float qAcc[64];

    if (threadIdx.x < 64) {
        sAcc[threadIdx.x] = 0.f; qAcc[threadIdx.x] = 0.f;
        if (BN_IN) {
            float a = 0.f, sh = 0.f;            // zero coeffs kill padded lanes
            if (threadIdx.x < FI) {
                const float m  = statsIn[threadIdx.x] * (1.0f / BATCH);
                const float vv = statsIn[64 + threadIdx.x] * (1.0f / BATCH) - m * m;
                a  = gam[threadIdx.x] * rsqrtf(vv + BN_EPS);   // biased var
                sh = bet[threadIdx.x] - m * a;
            }
            ssScale[threadIdx.x] = a;
            ssShift[threadIdx.x] = sh;
        }
    }
    __syncthreads();

    // --- weights -> registers in WMMA-B layout: K = (h&7) + 16*(h>>3) + 8*g + 32*kc, N = nloc
    v16h  breg[KC][NTT];
    float bias[NTT];
#pragma unroll
    for (int nt = 0; nt < NTT; ++nt) {
        const bool second = (nt >= NT1);
        const int  fo     = second ? FO2 : FO1;
        const float* W    = second ? W2 : W1;
        const float* B    = second ? B2 : B1;
        const int n = nloc + 16 * (second ? (nt - NT1) : nt);
        bias[nt] = (n < fo) ? B[n] : 0.0f;
#pragma unroll
        for (int kc = 0; kc < KC; ++kc) {
#pragma unroll
            for (int h = 0; h < 16; ++h) {
                const int k = (h & 7) + ((h >> 3) << 4) + 8 * g + 32 * kc;
                const float w = (n < fo && k < FI) ? W[n * FI + k] : 0.0f;
                breg[kc][nt][h] = (_Float16)w;
            }
        }
    }

    float sS[NT1], sQ[NT1];
#pragma unroll
    for (int nt = 0; nt < NT1; ++nt) { sS[nt] = 0.f; sQ[nt] = 0.f; }

    for (int tile = blockIdx.x * WAVES + wave; tile < TILES; tile += GRID * WAVES) {
        const int rowBase = tile * 16;
        const int M = nloc;   // A-matrix row owned by this lane

        // prefetch this wave's next tile (gfx1250 global_prefetch_b8)
        const int nextTile = tile + GRID * WAVES;
        if (nextTile < TILES) {
            if (IN_F32) __builtin_prefetch(zf + (size_t)(nextTile * 16 + M) * FI, 0, 1);
            else        __builtin_prefetch(zh + (size_t)(nextTile * 16 + M) * FIP, 0, 1);
        }

        v8f creg[NTT];
#pragma unroll
        for (int nt = 0; nt < NTT; ++nt)
#pragma unroll
            for (int j = 0; j < 8; ++j) creg[nt][j] = bias[nt];

        // lane-base pointers: all further offsets are compile-time immediates
        const _Float16* pz = zh + (size_t)(rowBase + M) * FIP + 8 * g;
        const _Float16* ps = SKIP_IN ? (skipIn + (size_t)(rowBase + M) * FIP + 8 * g)
                                     : nullptr;

#pragma unroll
        for (int kc = 0; kc < KC; ++kc) {
            // A layout (16-bit, 16x32): lane M = lane%16; two contiguous runs
            // per lane: K = 8g+32kc+{0..7} (h=0..7), K = 16+8g+32kc+{0..7} (h=8..15)
            v16h a;
            if (!IN_F32) {
#pragma unroll
                for (int run = 0; run < 2; ++run) {
                    const int k0 = 8 * g + 32 * kc + 16 * run;   // 8-half aligned
                    v8h r, sk;
#pragma unroll
                    for (int i = 0; i < 8; ++i) { r[i] = (_Float16)0; sk[i] = (_Float16)0; }
                    if (k0 < FI) {                         // EXEC-masked b128 load
                        r = *(const v8h*)(pz + 32 * kc + 16 * run);
                        if (SKIP_IN) sk = *(const v8h*)(ps + 32 * kc + 16 * run);
                    }
#pragma unroll
                    for (int i = 0; i < 8; ++i) {
                        const int k = k0 + i;
                        float v = (float)r[i];
                        if (BN_IN)   v = ssScale[k & 63] * v + ssShift[k & 63];
                        if (SKIP_IN) { v += (float)sk[i]; v = (k < FI) ? v : 0.0f; }
                        if (BN_IN)   v = fmaxf(v, 0.0f);   // relu follows every BN
                        a[8 * run + i] = (_Float16)v;
                    }
                }
            } else {   // stage 0: fp32 x, FI=5, scalar loads
#pragma unroll
                for (int h = 0; h < 16; ++h) {
                    const int k = (h & 7) + ((h >> 3) << 4) + 8 * g + 32 * kc;
                    float v = (k < FI) ? zf[(size_t)(rowBase + M) * FI + k] : 0.0f;
                    a[h] = (_Float16)v;
                }
            }
#pragma unroll
            for (int nt = 0; nt < NTT; ++nt)
                creg[nt] = __builtin_amdgcn_wmma_f32_16x16x32_f16(
                    false, a, false, breg[kc][nt], (short)0, creg[nt], false, false);
        }

        // --- store raw outputs (fp16, padded stride) + per-lane stats (fp32)
        //     base pointer per tile; j*fop folds into store immediate offsets
#pragma unroll
        for (int nt = 0; nt < NTT; ++nt) {
            const bool second = (nt >= NT1);
            const int  fo     = second ? FO2 : FO1;
            const int  fop    = second ? FOP2 : FOP1;
            const int  n      = nloc + 16 * (second ? (nt - NT1) : nt);
            if (n < fo) {
                _Float16* zo = second ? zout2 : zout1;
                _Float16* p  = zo + (size_t)(rowBase + 8 * g) * fop + n;
#pragma unroll
                for (int j = 0; j < 8; ++j) {
                    const float cv = creg[nt][j];                 // row rowBase+8g+j
                    p[j * fop] = (_Float16)cv;
                    if (!second) { sS[nt] += cv; sQ[nt] += cv * cv; }
                }
            }
        }
    }

    // --- block reduce stats via LDS float atomics, then one global add/feature
#pragma unroll
    for (int nt = 0; nt < NT1; ++nt) {
        const int n = nloc + 16 * nt;
        if (n < FO1) {
            atomicAdd(&sAcc[n], sS[nt]);
            atomicAdd(&qAcc[n], sQ[nt]);
        }
    }
    __syncthreads();
    if (threadIdx.x < FO1) {
        atomicAdd(&statsOut[threadIdx.x],      sAcc[threadIdx.x]);
        atomicAdd(&statsOut[64 + threadIdx.x], qAcc[threadIdx.x]);
    }
}

// ----------------------------------------------------------------------------
// Final: bn11 (2 features, stride pad8(2)=8) -> relu -> linear 2->1 (fp32 out)
// ----------------------------------------------------------------------------
__global__ __launch_bounds__(BLOCK) void final_kernel(
    const _Float16* __restrict__ z, const float* __restrict__ statsIn,
    const float* __restrict__ gam, const float* __restrict__ bet,
    const float* __restrict__ W, const float* __restrict__ B,
    float* __restrict__ out)
{
    const float m0 = statsIn[0] * (1.f / BATCH), m1 = statsIn[1] * (1.f / BATCH);
    const float v0 = statsIn[64] * (1.f / BATCH) - m0 * m0;
    const float v1 = statsIn[65] * (1.f / BATCH) - m1 * m1;
    const float a0 = gam[0] * rsqrtf(v0 + BN_EPS);
    const float a1 = gam[1] * rsqrtf(v1 + BN_EPS);
    const float s0 = bet[0] - m0 * a0, s1 = bet[1] - m1 * a1;
    const float w0 = W[0], w1 = W[1], b0 = B[0];
    for (int i = blockIdx.x * blockDim.x + threadIdx.x; i < BATCH;
         i += gridDim.x * blockDim.x) {
        const v2h p = *(const v2h*)(z + 8 * (size_t)i);    // aligned b32 pair
        const float h0 = fmaxf(a0 * (float)p[0] + s0, 0.f);
        const float h1 = fmaxf(a1 * (float)p[1] + s1, 0.f);
        out[i] = w0 * h0 + w1 * h1 + b0;
    }
}

// ----------------------------------------------------------------------------
extern "C" void kernel_launch(void* const* d_in, const int* in_sizes, int n_in,
                              void* d_out, int out_size, void* d_ws, size_t ws_size,
                              hipStream_t stream)
{
    (void)in_sizes; (void)n_in; (void)out_size; (void)ws_size;

    // setup_inputs order: x, (w1,b1..w12,b12), (sw1,sb1..sw3,sb3), (g1,be1..g11,be11)
    const float* x = (const float*)d_in[0];
    const float *w[13], *b[13], *sw[4], *sb[4], *gm[12], *be[12];
    for (int i = 1; i <= 12; ++i) {
        w[i] = (const float*)d_in[1 + 2 * (i - 1)];
        b[i] = (const float*)d_in[2 + 2 * (i - 1)];
    }
    for (int i = 1; i <= 3; ++i) {
        sw[i] = (const float*)d_in[25 + 2 * (i - 1)];
        sb[i] = (const float*)d_in[26 + 2 * (i - 1)];
    }
    for (int i = 1; i <= 11; ++i) {
        gm[i] = (const float*)d_in[31 + 2 * (i - 1)];
        be[i] = (const float*)d_in[32 + 2 * (i - 1)];
    }

    // workspace layout (fp16 activations, rows padded to 8-half multiples)
    char* ws = (char*)d_ws;
    const size_t N = BATCH;
    _Float16* zA = (_Float16*)ws; ws += N * 40 * sizeof(_Float16);   // 80 MB
    _Float16* zB = (_Float16*)ws; ws += N * 40 * sizeof(_Float16);   // 80 MB
    _Float16* s1 = (_Float16*)ws; ws += N * 24 * sizeof(_Float16);   // pad8(20)
    _Float16* s2 = (_Float16*)ws; ws += N * 24 * sizeof(_Float16);
    _Float16* s3 = (_Float16*)ws; ws += N * 8  * sizeof(_Float16);   // pad8(5)
    float* stats = (float*)ws;    // 11 slots * 128 floats (sum[64] + sumsq[64])

    hipMemsetAsync(stats, 0, 11 * 128 * sizeof(float), stream);

    dim3 gr(GRID), bl(BLOCK);
#define ST(i) (stats + (i) * 128)
    // K0: x -> z1p (w1) + s1 (sw1)                          [stats for bn1]
    stage_kernel<5, 10, 20, false, false, true><<<gr, bl, 0, stream>>>(
        x, nullptr, nullptr, nullptr, nullptr, w[1], b[1], sw[1], sb[1], zA, s1, ST(0));
    // K1: bn1(z1p)->relu -> z2p (w2)
    stage_kernel<10, 10, 0, true, false, false><<<gr, bl, 0, stream>>>(
        zA, ST(0), gm[1], be[1], nullptr, w[2], b[2], nullptr, nullptr, zB, nullptr, ST(1));
    // K2: bn2 -> z3p (w3)
    stage_kernel<10, 20, 0, true, false, false><<<gr, bl, 0, stream>>>(
        zB, ST(1), gm[2], be[2], nullptr, w[3], b[3], nullptr, nullptr, zA, nullptr, ST(2));
    // K3: relu(bn3+s1)=z3 -> z4p (w4) + s2 (sw2)
    stage_kernel<20, 40, 20, true, true, false><<<gr, bl, 0, stream>>>(
        zA, ST(2), gm[3], be[3], s1, w[4], b[4], sw[2], sb[2], zB, s2, ST(3));
    // K4: bn4 -> z5p (w5)
    stage_kernel<40, 40, 0, true, false, false><<<gr, bl, 0, stream>>>(
        zB, ST(3), gm[4], be[4], nullptr, w[5], b[5], nullptr, nullptr, zA, nullptr, ST(4));
    // K5: bn5 -> z6p (w6)
    stage_kernel<40, 20, 0, true, false, false><<<gr, bl, 0, stream>>>(
        zA, ST(4), gm[5], be[5], nullptr, w[6], b[6], nullptr, nullptr, zB, nullptr, ST(5));
    // K6: relu(bn6+s2)=z6 -> z7p (w7) + s3 (sw3)
    stage_kernel<20, 10, 5, true, true, false><<<gr, bl, 0, stream>>>(
        zB, ST(5), gm[6], be[6], s2, w[7], b[7], sw[3], sb[3], zA, s3, ST(6));
    // K7: bn7 -> z8p (w8)
    stage_kernel<10, 10, 0, true, false, false><<<gr, bl, 0, stream>>>(
        zA, ST(6), gm[7], be[7], nullptr, w[8], b[8], nullptr, nullptr, zB, nullptr, ST(7));
    // K8: bn8 -> z9p (w9)
    stage_kernel<10, 5, 0, true, false, false><<<gr, bl, 0, stream>>>(
        zB, ST(7), gm[8], be[8], nullptr, w[9], b[9], nullptr, nullptr, zA, nullptr, ST(8));
    // K9: relu(bn9+s3)=z9 -> z10p (w10)
    stage_kernel<5, 2, 0, true, true, false><<<gr, bl, 0, stream>>>(
        zA, ST(8), gm[9], be[9], s3, w[10], b[10], nullptr, nullptr, zB, nullptr, ST(9));
    // K10: bn10 -> z11p (w11)
    stage_kernel<2, 2, 0, true, false, false><<<gr, bl, 0, stream>>>(
        zB, ST(9), gm[10], be[10], nullptr, w[11], b[11], nullptr, nullptr, zA, nullptr, ST(10));
    // K11: bn11 -> relu -> linear 2->1
    final_kernel<<<gr, bl, 0, stream>>>(zA, ST(10), gm[11], be[11], w[12], b[12],
                                        (float*)d_out);
#undef ST
}
